// AttentionMoeModel_40733469835726
// MI455X (gfx1250) — compile-verified, matrix operand order checked by reading, not run on hardware
//
#include <hip/hip_runtime.h>

// ---------------- model constants ----------------
#define T_  1024
#define D_  1024
#define H_  8
#define HD_ 128
#define V_  32000
#define L_  4
#define E_  8
#define F_  1024

// ---------------- WMMA types ----------------
typedef __attribute__((ext_vector_type(16))) __bf16 v16bf;
typedef __attribute__((ext_vector_type(8)))  float  v8f;
typedef int i32x4 __attribute__((vector_size(16)));

union FragAB { v16bf v; unsigned int u[8]; };

#define AS1 __attribute__((address_space(1)))
#define AS3 __attribute__((address_space(3)))

#if defined(__gfx1250__) && __has_builtin(__builtin_amdgcn_global_load_async_to_lds_b128)
#define USE_ASYNC 1
#else
#define USE_ASYNC 0
#endif

__device__ __forceinline__ void wait_async0() {
#if USE_ASYNC
#if __has_builtin(__builtin_amdgcn_s_wait_asynccnt)
  __builtin_amdgcn_s_wait_asynccnt(0);
#else
  asm volatile("s_wait_asynccnt 0" ::: "memory");
#endif
#endif
}

__device__ __forceinline__ void cp16(const unsigned short* src, unsigned short* ldsDst) {
#if USE_ASYNC
  __builtin_amdgcn_global_load_async_to_lds_b128((AS1 i32x4*)src, (AS3 i32x4*)ldsDst, 0, 0);
#else
  __builtin_prefetch(src, 0, 1);
  *(uint4*)ldsDst = *(const uint4*)src;
#endif
}

__device__ __forceinline__ unsigned short f2bf(float f) {
  unsigned int u = __float_as_uint(f);
  u += 0x7FFFu + ((u >> 16) & 1u);   // round-to-nearest-even
  return (unsigned short)(u >> 16);
}

__device__ __forceinline__ float bredsum(float v, float* sred) {
#pragma unroll
  for (int o = 16; o > 0; o >>= 1) v += __shfl_xor(v, o, 32);
  if ((threadIdx.x & 31) == 0) sred[threadIdx.x >> 5] = v;
  __syncthreads();
  float tot = 0.f;
  int nw = blockDim.x >> 5;
  for (int i = 0; i < nw; i++) tot += sred[i];
  __syncthreads();
  return tot;
}

__device__ __forceinline__ float bredmax(float v, float* sred) {
#pragma unroll
  for (int o = 16; o > 0; o >>= 1) v = fmaxf(v, __shfl_xor(v, o, 32));
  if ((threadIdx.x & 31) == 0) sred[threadIdx.x >> 5] = v;
  __syncthreads();
  float tot = -3.0e38f;
  int nw = blockDim.x >> 5;
  for (int i = 0; i < nw; i++) tot = fmaxf(tot, sred[i]);
  __syncthreads();
  return tot;
}

// ---------------- generic bf16 WMMA GEMM ----------------
// C[m][n] (+)= alpha * sum_k A[m][k] * B[n][k]   (B stored N-major, k contiguous)
// Optional per-batch row count / base offset, A row gather, C row scatter (MoE).
// Block: 128 threads (4 waves, 2x2), tile 64(M) x 128(N) x 64(K).
// Each wave: 32x64 C tile = 2x4 WMMA frags; K consumed as 2 sub-steps of 32.
#define TM 64
#define TN 128
#define TK 64

__global__ __launch_bounds__(128)
void gemm_bf16(const unsigned short* __restrict__ A,
               const unsigned short* __restrict__ B,
               float* __restrict__ C,
               int M, int N, int Kd,
               int lda, int ldb, int ldc,
               long long sA, long long sB, long long sC,
               int accum, float alpha,
               const int* __restrict__ mCount,
               const int* __restrict__ mBase,
               const int* __restrict__ aGather,
               const int* __restrict__ cScatter)
{
  int z = blockIdx.z;
  A += (long long)z * sA;
  B += (long long)z * sB;
  C += (long long)z * sC;
  int mEff = mCount ? mCount[z] : M;
  int base = mBase ? mBase[z] : 0;
  int m0 = blockIdx.y * TM;
  int n0 = blockIdx.x * TN;
  if (m0 >= mEff || n0 >= N) return;

  __shared__ alignas(16) unsigned short As[TM][TK];  // [m][k]  8 KB
  __shared__ alignas(16) unsigned short Bs[TN][TK];  // [n][k] 16 KB

  int tid  = threadIdx.x;
  int wave = tid >> 5;
  int lane = tid & 31;
  int hi   = lane >> 4;
  int lxy  = lane & 15;
  int wm   = (wave >> 1) * 32;
  int wn   = (wave & 1) * 64;

  v8f acc[2][4];
#pragma unroll
  for (int i = 0; i < 2; i++)
#pragma unroll
    for (int j = 0; j < 4; j++)
#pragma unroll
      for (int r = 0; r < 8; r++) acc[i][j][r] = 0.0f;

  for (int k0 = 0; k0 < Kd; k0 += TK) {
    // stage A tile: 64 rows x 64 k = 512 chunks of 16B
    for (int c = tid; c < (TM * TK) / 8; c += 128) {
      int m    = c >> 3;
      int koff = (c & 7) * 8;
      int gm   = m0 + m;
      if (gm < mEff) {
        int arow = aGather ? aGather[base + gm] : (base + gm);
        cp16(A + (long long)arow * lda + k0 + koff, &As[m][koff]);
      } else {
        *(uint4*)&As[m][koff] = make_uint4(0, 0, 0, 0);
      }
    }
    // stage B tile: 128 rows x 64 k = 1024 chunks of 16B
    for (int c = tid; c < (TN * TK) / 8; c += 128) {
      int n    = c >> 3;
      int koff = (c & 7) * 8;
      int gn   = n0 + n;
      if (gn < N) {
        cp16(B + (long long)gn * ldb + k0 + koff, &Bs[n][koff]);
      } else {
        *(uint4*)&Bs[n][koff] = make_uint4(0, 0, 0, 0);
      }
    }
    wait_async0();
    __syncthreads();

#pragma unroll
    for (int s = 0; s < 2; s++) {
      FragAB bfr[4];
#pragma unroll
      for (int j = 0; j < 4; j++) {
#pragma unroll
        for (int v = 0; v < 8; v++) {
          int kk = s * 32 + ((v >> 2) << 4) + (hi << 3) + ((v & 3) << 1);
          bfr[j].u[v] = *(const unsigned int*)&Bs[wn + j * 16 + lxy][kk];
        }
      }
#pragma unroll
      for (int i = 0; i < 2; i++) {
        FragAB afr;
#pragma unroll
        for (int v = 0; v < 8; v++) {
          int kk = s * 32 + ((v >> 2) << 4) + (hi << 3) + ((v & 3) << 1);
          afr.u[v] = *(const unsigned int*)&As[wm + i * 16 + lxy][kk];
        }
#pragma unroll
        for (int j = 0; j < 4; j++) {
          acc[i][j] = __builtin_amdgcn_wmma_f32_16x16x32_bf16(
              false, afr.v, false, bfr[j].v, (short)0, acc[i][j], false, false);
        }
      }
    }
    __syncthreads();
  }

  // C tile: lane holds column n=lxy (half hi), VGPR r -> row r + 8*hi
#pragma unroll
  for (int i = 0; i < 2; i++) {
#pragma unroll
    for (int j = 0; j < 4; j++) {
      int nb = n0 + wn + j * 16 + lxy;
      if (nb >= N) continue;
#pragma unroll
      for (int r = 0; r < 8; r++) {
        int mb = m0 + wm + i * 16 + r + hi * 8;
        if (mb >= mEff) continue;
        int crow = cScatter ? cScatter[base + mb] : (base + mb);
        float val = alpha * acc[i][j][r];
        float* dst = C + (long long)crow * ldc + nb;
        if (accum) *dst += val; else *dst = val;
      }
    }
  }
}

// ---------------- elementwise / reduction kernels ----------------

__global__ void k_f32_to_bf16(const float* __restrict__ s,
                              unsigned short* __restrict__ d, long long n) {
  long long i = (long long)blockIdx.x * blockDim.x + threadIdx.x;
  long long stride = (long long)gridDim.x * blockDim.x;
  for (; i < n; i += stride) d[i] = f2bf(s[i]);
}

// dst[b][n][k] = bf16(src[b][k][n])   (weights -> N-major bf16)
__global__ void k_transpose_bf16(const float* __restrict__ s,
                                 unsigned short* __restrict__ d,
                                 int K, int N, long long total) {
  long long i = (long long)blockIdx.x * blockDim.x + threadIdx.x;
  long long stride = (long long)gridDim.x * blockDim.x;
  long long kn = (long long)K * N;
  for (; i < total; i += stride) {
    long long b = i / kn;
    long long r = i % kn;
    int k = (int)(r / N);
    int n = (int)(r % N);
    d[b * kn + (long long)n * K + k] = f2bf(s[i]);
  }
}

// x = x0 = rms(wte[idx[t]])
__global__ __launch_bounds__(256)
void k_embed(const int* __restrict__ idx, const float* __restrict__ wte,
             float* __restrict__ x, float* __restrict__ x0) {
  __shared__ float sred[8];
  int t = blockIdx.x;
  const float* w = wte + (long long)idx[t] * D_;
  float vals[4];
  float ss = 0.f;
#pragma unroll
  for (int j = 0; j < 4; j++) {
    vals[j] = w[threadIdx.x + 256 * j];
    ss += vals[j] * vals[j];
  }
  float tot = bredsum(ss, sred);
  float sc = rsqrtf(tot / (float)D_ + 1e-6f);
#pragma unroll
  for (int j = 0; j < 4; j++) {
    float v = vals[j] * sc;
    x [(long long)t * D_ + threadIdx.x + 256 * j] = v;
    x0[(long long)t * D_ + threadIdx.x + 256 * j] = v;
  }
}

__global__ void k_resid(float* __restrict__ x, const float* __restrict__ x0,
                        const float* __restrict__ rl, const float* __restrict__ xl,
                        int layer, long long n) {
  float a = rl[layer], b = xl[layer];
  long long i = (long long)blockIdx.x * blockDim.x + threadIdx.x;
  long long stride = (long long)gridDim.x * blockDim.x;
  for (; i < n; i += stride) x[i] = a * x[i] + b * x0[i];
}

// xn = rms(x) (f32 + bf16)
__global__ __launch_bounds__(256)
void k_rmsnorm(const float* __restrict__ x, float* __restrict__ xn,
               unsigned short* __restrict__ xnb) {
  __shared__ float sred[8];
  int t = blockIdx.x;
  const float* r = x + (long long)t * D_;
  float vals[4];
  float ss = 0.f;
#pragma unroll
  for (int j = 0; j < 4; j++) {
    vals[j] = r[threadIdx.x + 256 * j];
    ss += vals[j] * vals[j];
  }
  float tot = bredsum(ss, sred);
  float sc = rsqrtf(tot / (float)D_ + 1e-6f);
#pragma unroll
  for (int j = 0; j < 4; j++) {
    float v = vals[j] * sc;
    long long o = (long long)t * D_ + threadIdx.x + 256 * j;
    if (xn)  xn[o]  = v;
    if (xnb) xnb[o] = f2bf(v);
  }
}

// RoPE + per-head RMS: grid = T_*H_ blocks of 128 threads
__global__ __launch_bounds__(128)
void k_rope_rms(const float* __restrict__ X, unsigned short* __restrict__ Ob) {
  __shared__ float sred[4];
  int th = blockIdx.x;
  int t = th / H_;
  int h = th % H_;
  const float* src = X + (long long)t * D_ + h * HD_;
  int d = threadIdx.x;
  int i = d & 63;
  float x1 = src[i];
  float x2 = src[64 + i];
  float inv = __powf(10000.0f, -((float)(2 * i)) / (float)HD_);
  float fr = (float)t * inv;
  float c = __cosf(fr), s = __sinf(fr);
  float val = (d < 64) ? (x1 * c + x2 * s) : (-x1 * s + x2 * c);
  float ss = val * val;
#pragma unroll
  for (int o = 16; o > 0; o >>= 1) ss += __shfl_xor(ss, o, 32);
  if ((threadIdx.x & 31) == 0) sred[threadIdx.x >> 5] = ss;
  __syncthreads();
  float tot = sred[0] + sred[1] + sred[2] + sred[3];
  float sc = rsqrtf(tot / (float)HD_ + 1e-6f);
  Ob[(long long)t * D_ + h * HD_ + d] = f2bf(val * sc);
}

// gates[t*H+h] = 2*sigmoid(xn[t,:32] @ vgw[:,h])
__global__ void k_vegate(const float* __restrict__ xn, const float* __restrict__ vgw,
                         float* __restrict__ gates) {
  int i = blockIdx.x * blockDim.x + threadIdx.x;
  if (i >= T_ * H_) return;
  int t = i / H_, h = i % H_;
  float acc = 0.f;
  for (int c = 0; c < 32; c++) acc += xn[(long long)t * D_ + c] * vgw[c * H_ + h];
  gates[i] = 2.0f / (1.0f + __expf(-acc));
}

// vbT[cc][t] = bf16(v[t][cc] + gate*ve)   (transposed for PV GEMM B-side)
__global__ void k_vprep(const float* __restrict__ v, const float* __restrict__ gates,
                        const float* __restrict__ ve, const int* __restrict__ idx,
                        unsigned short* __restrict__ vbT, int hasVE) {
  long long i = (long long)blockIdx.x * blockDim.x + threadIdx.x;
  long long n = (long long)T_ * D_;
  long long stride = (long long)gridDim.x * blockDim.x;
  for (; i < n; i += stride) {
    int t = (int)(i / D_);
    int cc = (int)(i % D_);
    float val = v[i];
    if (hasVE) {
      int h = cc / HD_;
      val += gates[t * H_ + h] * ve[(long long)idx[t] * D_ + cc];
    }
    vbT[(long long)cc * T_ + t] = f2bf(val);
  }
}

// masked softmax over key dim; S f32 -> P bf16. grid (T_, H_), 256 threads
__global__ __launch_bounds__(256)
void k_softmax(const float* __restrict__ S, unsigned short* __restrict__ P, int window) {
  __shared__ float sred[8];
  int q = blockIdx.x;
  int h = blockIdx.y;
  const float* row = S + ((long long)h * T_ + q) * T_;
  unsigned short* prow = P + ((long long)h * T_ + q) * T_;
  float mx = -3.0e38f;
  for (int j = threadIdx.x; j < T_; j += 256) {
    bool ok = (j <= q) && (q - j <= window);
    float vv = ok ? row[j] : -3.0e38f;
    mx = fmaxf(mx, vv);
  }
  mx = bredmax(mx, sred);
  float sum = 0.f;
  for (int j = threadIdx.x; j < T_; j += 256) {
    bool ok = (j <= q) && (q - j <= window);
    if (ok) sum += __expf(row[j] - mx);
  }
  sum = bredsum(sum, sred);
  float inv = 1.0f / sum;
  for (int j = threadIdx.x; j < T_; j += 256) {
    bool ok = (j <= q) && (q - j <= window);
    float pv = ok ? __expf(row[j] - mx) * inv : 0.0f;
    prow[j] = f2bf(pv);
  }
}

__global__ void k_relu2(const float* __restrict__ h, unsigned short* __restrict__ hb,
                        long long n) {
  long long i = (long long)blockIdx.x * blockDim.x + threadIdx.x;
  long long stride = (long long)gridDim.x * blockDim.x;
  for (; i < n; i += stride) {
    float r = fmaxf(h[i], 0.0f);
    hb[i] = f2bf(r * r);
  }
}

__global__ void k_silu(const float* __restrict__ h, unsigned short* __restrict__ hb,
                       long long n) {
  long long i = (long long)blockIdx.x * blockDim.x + threadIdx.x;
  long long stride = (long long)gridDim.x * blockDim.x;
  for (; i < n; i += stride) {
    float v = h[i];
    hb[i] = f2bf(v / (1.0f + __expf(-v)));
  }
}

// sact[t][f] = bf16(silu(gu[t][f]) * gu[t][F+f])
__global__ void k_sharedact(const float* __restrict__ gu, unsigned short* __restrict__ sact) {
  long long i = (long long)blockIdx.x * blockDim.x + threadIdx.x;
  long long n = (long long)T_ * F_;
  long long stride = (long long)gridDim.x * blockDim.x;
  for (; i < n; i += stride) {
    int t = (int)(i / F_);
    int f = (int)(i % F_);
    float g = gu[(long long)t * (2 * F_) + f];
    float u = gu[(long long)t * (2 * F_) + F_ + f];
    sact[i] = f2bf((g / (1.0f + __expf(-g))) * u);
  }
}

// router: softmax(xn @ rw) then top-2; one thread per token
__global__ void k_router(const float* __restrict__ xn, const float* __restrict__ rw,
                         float* __restrict__ tw, int* __restrict__ ti) {
  int t = blockIdx.x * blockDim.x + threadIdx.x;
  if (t >= T_) return;
  const float* xr = xn + (long long)t * D_;
  float logits[E_];
#pragma unroll
  for (int e = 0; e < E_; e++) logits[e] = 0.f;
  for (int d = 0; d < D_; d++) {
    float xv = xr[d];
#pragma unroll
    for (int e = 0; e < E_; e++) logits[e] += xv * rw[d * E_ + e];
  }
  float mx = logits[0];
#pragma unroll
  for (int e = 1; e < E_; e++) mx = fmaxf(mx, logits[e]);
  float sum = 0.f;
#pragma unroll
  for (int e = 0; e < E_; e++) { logits[e] = __expf(logits[e] - mx); sum += logits[e]; }
  float inv = 1.0f / sum;
#pragma unroll
  for (int e = 0; e < E_; e++) logits[e] *= inv;
  int i0 = 0;
#pragma unroll
  for (int e = 1; e < E_; e++) if (logits[e] > logits[i0]) i0 = e;
  int i1 = (i0 == 0) ? 1 : 0;
#pragma unroll
  for (int e = 0; e < E_; e++) if (e != i0 && logits[e] > logits[i1]) i1 = e;
  ti[t * 2 + 0] = i0; tw[t * 2 + 0] = logits[i0];
  ti[t * 2 + 1] = i1; tw[t * 2 + 1] = logits[i1];
}

__global__ void k_zero_counts(int* counts) {
  if (threadIdx.x < E_) counts[threadIdx.x] = 0;
}

__global__ void k_count(const int* __restrict__ ti, int* __restrict__ counts) {
  int i = blockIdx.x * blockDim.x + threadIdx.x;
  if (i < T_ * 2) atomicAdd(&counts[ti[i]], 1);
}

__global__ void k_offsets(const int* __restrict__ counts, int* __restrict__ offsets,
                          int* __restrict__ cursor) {
  if (threadIdx.x == 0 && blockIdx.x == 0) {
    int acc = 0;
    for (int e = 0; e < E_; e++) { offsets[e] = acc; acc += counts[e]; cursor[e] = 0; }
  }
}

__global__ void k_place(const int* __restrict__ ti, const int* __restrict__ offsets,
                        int* __restrict__ cursor, int* __restrict__ rowIdx,
                        int* __restrict__ slotIds) {
  int i = blockIdx.x * blockDim.x + threadIdx.x;
  if (i >= T_ * 2) return;
  int e = ti[i];
  int pos = atomicAdd(&cursor[e], 1);
  int dst = offsets[e] + pos;
  rowIdx[dst] = i >> 1;   // token
  slotIds[dst] = i;       // slot = t*2 + s
}

__global__ void k_sgate(const float* __restrict__ xn, const float* __restrict__ gw,
                        float* __restrict__ sg) {
  int t = blockIdx.x * blockDim.x + threadIdx.x;
  if (t >= T_) return;
  float acc = 0.f;
  const float* xr = xn + (long long)t * D_;
  for (int d = 0; d < D_; d++) acc += xr[d] * gw[d];
  sg[t] = 1.0f / (1.0f + __expf(-acc));
}

// x += tw0*slot0 + tw1*slot1 + sg*sharedTmp
__global__ void k_combine(float* __restrict__ x, const float* __restrict__ tw,
                          const float* __restrict__ slotOut,
                          const float* __restrict__ sharedTmp,
                          const float* __restrict__ sg) {
  long long i = (long long)blockIdx.x * blockDim.x + threadIdx.x;
  long long n = (long long)T_ * D_;
  long long stride = (long long)gridDim.x * blockDim.x;
  for (; i < n; i += stride) {
    int t = (int)(i / D_);
    int dcol = (int)(i % D_);
    float r = tw[t * 2 + 0] * slotOut[(long long)(t * 2 + 0) * D_ + dcol]
            + tw[t * 2 + 1] * slotOut[(long long)(t * 2 + 1) * D_ + dcol]
            + sg[t] * sharedTmp[i];
    x[i] += r;
  }
}

// ---------------- host orchestration ----------------
extern "C" void kernel_launch(void* const* d_in, const int* in_sizes, int n_in,
                              void* d_out, int out_size, void* d_ws, size_t ws_size,
                              hipStream_t stream) {
  (void)in_sizes; (void)n_in; (void)out_size; (void)ws_size;

  const int*   idx    = (const int*)  d_in[0];
  const float* wte    = (const float*)d_in[1];
  const float* lmw    = (const float*)d_in[2];
  const float* wq     = (const float*)d_in[3];
  const float* wk     = (const float*)d_in[4];
  const float* wv     = (const float*)d_in[5];
  const float* wo     = (const float*)d_in[6];
  const float* veT    = (const float*)d_in[7];
  const float* vgw    = (const float*)d_in[8];
  const float* fc     = (const float*)d_in[9];
  const float* proj   = (const float*)d_in[10];
  const float* rw     = (const float*)d_in[11];
  const float* w1     = (const float*)d_in[12];
  const float* w2     = (const float*)d_in[13];
  const float* guw    = (const float*)d_in[14];
  const float* downw  = (const float*)d_in[15];
  const float* gatew  = (const float*)d_in[16];
  const float* rl     = (const float*)d_in[17];
  const float* xl     = (const float*)d_in[18];

  char* pws = (char*)d_ws;
  auto alloc = [&](size_t bytes) -> void* {
    void* r = (void*)pws;
    pws += (bytes + 255) & ~(size_t)255;
    return r;
  };

  const long long DD = (long long)D_ * D_;
  // bf16 weights, stored TRANSPOSED (N-major, k contiguous) for the GEMM B side
  unsigned short* wqb   = (unsigned short*)alloc((size_t)L_ * DD * 2);
  unsigned short* wkb   = (unsigned short*)alloc((size_t)L_ * DD * 2);
  unsigned short* wvb   = (unsigned short*)alloc((size_t)L_ * DD * 2);
  unsigned short* wob   = (unsigned short*)alloc((size_t)L_ * DD * 2);
  unsigned short* fcb   = (unsigned short*)alloc((size_t)2 * D_ * 4 * D_ * 2);
  unsigned short* projb = (unsigned short*)alloc((size_t)2 * 4 * D_ * D_ * 2);
  unsigned short* w1b   = (unsigned short*)alloc((size_t)2 * E_ * D_ * F_ * 2);
  unsigned short* w2b   = (unsigned short*)alloc((size_t)2 * E_ * F_ * D_ * 2);
  unsigned short* gub   = (unsigned short*)alloc((size_t)2 * D_ * 2 * F_ * 2);
  unsigned short* downb = (unsigned short*)alloc((size_t)2 * F_ * D_ * 2);
  unsigned short* lmb   = (unsigned short*)alloc((size_t)D_ * V_ * 2);
  // activations
  float* x    = (float*)alloc((size_t)T_ * D_ * 4);
  float* x0   = (float*)alloc((size_t)T_ * D_ * 4);
  float* xn   = (float*)alloc((size_t)T_ * D_ * 4);
  unsigned short* xnb = (unsigned short*)alloc((size_t)T_ * D_ * 2);
  float* qf = (float*)alloc((size_t)T_ * D_ * 4);
  float* kf = (float*)alloc((size_t)T_ * D_ * 4);
  float* vf = (float*)alloc((size_t)T_ * D_ * 4);
  unsigned short* qb  = (unsigned short*)alloc((size_t)T_ * D_ * 2);
  unsigned short* kb  = (unsigned short*)alloc((size_t)T_ * D_ * 2);
  unsigned short* vbT = (unsigned short*)alloc((size_t)T_ * D_ * 2);
  float* yf = (float*)alloc((size_t)T_ * D_ * 4);
  unsigned short* yb = (unsigned short*)alloc((size_t)T_ * D_ * 2);
  float* S  = (float*)alloc((size_t)H_ * T_ * T_ * 4);
  unsigned short* Pb = (unsigned short*)alloc((size_t)H_ * T_ * T_ * 2);
  float* hbuf = (float*)alloc((size_t)T_ * 4 * D_ * 4);
  unsigned short* hb = (unsigned short*)alloc((size_t)T_ * 4 * D_ * 2);
  float* gu   = (float*)alloc((size_t)T_ * 2 * F_ * 4);
  unsigned short* sact = (unsigned short*)alloc((size_t)T_ * F_ * 2);
  float* sharedTmp = (float*)alloc((size_t)T_ * D_ * 4);
  float* moeH = (float*)alloc((size_t)2 * T_ * F_ * 4);
  unsigned short* moeHb = (unsigned short*)alloc((size_t)2 * T_ * F_ * 2);
  float* slotOut = (float*)alloc((size_t)2 * T_ * D_ * 4);
  float* tw   = (float*)alloc((size_t)T_ * 2 * 4);
  int*   ti   = (int*)alloc((size_t)T_ * 2 * 4);
  int* counts  = (int*)alloc(E_ * 4);
  int* offsets = (int*)alloc(E_ * 4);
  int* cursor  = (int*)alloc(E_ * 4);
  int* rowIdx  = (int*)alloc((size_t)T_ * 2 * 4);
  int* slotIds = (int*)alloc((size_t)T_ * 2 * 4);
  float* gates = (float*)alloc((size_t)T_ * H_ * 4);
  float* sg    = (float*)alloc((size_t)T_ * 4);

  auto conv = [&](const float* s, unsigned short* d, long long n) {
    long long blocks = (n + 255) / 256;
    if (blocks > 32768) blocks = 32768;
    k_f32_to_bf16<<<dim3((unsigned)blocks), dim3(256), 0, stream>>>(s, d, n);
  };
  auto convT = [&](const float* s, unsigned short* d, int K, int N, int batch) {
    long long n = (long long)batch * K * N;
    long long blocks = (n + 255) / 256;
    if (blocks > 32768) blocks = 32768;
    k_transpose_bf16<<<dim3((unsigned)blocks), dim3(256), 0, stream>>>(s, d, K, N, n);
  };
  auto gemm = [&](const unsigned short* A, const unsigned short* B, float* C,
                  int M, int N, int K, int lda, int ldb, int ldc,
                  long long sA, long long sB, long long sC, int batch,
                  int acc, float alpha,
                  const int* mc, const int* mb, const int* ag, const int* cs) {
    dim3 g((N + TN - 1) / TN, (M + TM - 1) / TM, batch);
    gemm_bf16<<<g, dim3(128), 0, stream>>>(A, B, C, M, N, K, lda, ldb, ldc,
                                           sA, sB, sC, acc, alpha, mc, mb, ag, cs);
  };

  // convert+transpose all weights to bf16 N-major
  convT(wq,    wqb,   D_,     D_,     L_);
  convT(wk,    wkb,   D_,     D_,     L_);
  convT(wv,    wvb,   D_,     D_,     L_);
  convT(wo,    wob,   D_,     D_,     L_);
  convT(fc,    fcb,   D_,     4 * D_, 2);
  convT(proj,  projb, 4 * D_, D_,     2);
  convT(w1,    w1b,   D_,     F_,     2 * E_);
  convT(w2,    w2b,   F_,     D_,     2 * E_);
  convT(guw,   gub,   D_,     2 * F_, 2);
  convT(downw, downb, F_,     D_,     2);
  convT(lmw,   lmb,   D_,     V_,     1);

  const int windows[L_] = {1024, 512, 1024, 1024};
  const long long TD = (long long)T_ * D_;

  // embed + rms
  k_embed<<<dim3(T_), dim3(256), 0, stream>>>(idx, wte, x, x0);

  for (int layer = 0; layer < L_; layer++) {
    // x = rl*x + xl*x0
    k_resid<<<dim3(4096), dim3(256), 0, stream>>>(x, x0, rl, xl, layer, TD);
    // xn = rms(x)
    k_rmsnorm<<<dim3(T_), dim3(256), 0, stream>>>(x, xn, xnb);

    // q,k,v projections (B = transposed weight, ldb = K)
    gemm(xnb, wqb + (long long)layer * DD, qf, T_, D_, D_, D_, D_, D_,
         0, 0, 0, 1, 0, 1.0f, nullptr, nullptr, nullptr, nullptr);
    gemm(xnb, wkb + (long long)layer * DD, kf, T_, D_, D_, D_, D_, D_,
         0, 0, 0, 1, 0, 1.0f, nullptr, nullptr, nullptr, nullptr);
    gemm(xnb, wvb + (long long)layer * DD, vf, T_, D_, D_, D_, D_, D_,
         0, 0, 0, 1, 0, 1.0f, nullptr, nullptr, nullptr, nullptr);

    // value embeddings on layers 0 and 3
    int hasVE = (layer == 0 || layer == 3) ? 1 : 0;
    if (hasVE) {
      int j = (layer == 0) ? 0 : 1;
      k_vegate<<<dim3((T_ * H_ + 255) / 256), dim3(256), 0, stream>>>(
          xn, vgw + (long long)j * 32 * H_, gates);
      k_vprep<<<dim3(4096), dim3(256), 0, stream>>>(
          vf, gates, veT + (long long)j * V_ * D_, idx, vbT, 1);
    } else {
      k_vprep<<<dim3(4096), dim3(256), 0, stream>>>(vf, gates, veT, idx, vbT, 0);
    }

    // rope + rms on q, k
    k_rope_rms<<<dim3(T_ * H_), dim3(128), 0, stream>>>(qf, qb);
    k_rope_rms<<<dim3(T_ * H_), dim3(128), 0, stream>>>(kf, kb);

    // scores: S[h] = (Qh @ Kh^T) / sqrt(HD); K already N-major (token rows)
    gemm(qb, kb, S, T_, T_, HD_, D_, D_, T_,
         HD_, HD_, (long long)T_ * T_, H_, 0, 0.08838834764831845f,
         nullptr, nullptr, nullptr, nullptr);
    // masked softmax -> P (bf16)
    k_softmax<<<dim3(T_, H_), dim3(256), 0, stream>>>(S, Pb, windows[layer]);
    // y[h] = P[h] @ Vh   (V pre-transposed per head: vbT[(h*HD+n)*T + t])
    gemm(Pb, vbT, yf, T_, HD_, T_, T_, T_, D_,
         (long long)T_ * T_, (long long)HD_ * T_, HD_, H_, 0, 1.0f,
         nullptr, nullptr, nullptr, nullptr);
    // o-proj, accumulate into x
    conv(yf, yb, TD);
    gemm(yb, wob + (long long)layer * DD, x, T_, D_, D_, D_, D_, D_,
         0, 0, 0, 1, 1, 1.0f, nullptr, nullptr, nullptr, nullptr);

    // MLP / MoE
    k_rmsnorm<<<dim3(T_), dim3(256), 0, stream>>>(x, xn, xnb);
    if (layer < 2) {
      // dense: x += relu(xn@fc)^2 @ proj
      gemm(xnb, fcb + (long long)layer * D_ * 4 * D_, hbuf, T_, 4 * D_, D_,
           D_, D_, 4 * D_, 0, 0, 0, 1, 0, 1.0f,
           nullptr, nullptr, nullptr, nullptr);
      k_relu2<<<dim3(16384), dim3(256), 0, stream>>>(hbuf, hb, (long long)T_ * 4 * D_);
      gemm(hb, projb + (long long)layer * 4 * D_ * D_, x, T_, D_, 4 * D_,
           4 * D_, 4 * D_, D_, 0, 0, 0, 1, 1, 1.0f,
           nullptr, nullptr, nullptr, nullptr);
    } else {
      int j = layer - 2;
      // routing
      k_router<<<dim3((T_ + 255) / 256), dim3(256), 0, stream>>>(
          xn, rw + (long long)j * D_ * E_, tw, ti);
      k_zero_counts<<<dim3(1), dim3(32), 0, stream>>>(counts);
      k_count<<<dim3((T_ * 2 + 255) / 256), dim3(256), 0, stream>>>(ti, counts);
      k_offsets<<<dim3(1), dim3(32), 0, stream>>>(counts, offsets, cursor);
      k_place<<<dim3((T_ * 2 + 255) / 256), dim3(256), 0, stream>>>(
          ti, offsets, cursor, rowIdx, slotIds);
      // expert GEMM 1: moeH[rows] = gather(xn) @ w1[e]
      gemm(xnb, w1b + (long long)j * E_ * D_ * F_, moeH,
           2 * T_, F_, D_, D_, D_, F_,
           0, (long long)D_ * F_, 0, E_, 0, 1.0f,
           counts, offsets, rowIdx, nullptr);
      k_silu<<<dim3(8192), dim3(256), 0, stream>>>(moeH, moeHb, (long long)2 * T_ * F_);
      // expert GEMM 2: slotOut[slot] = silu(h) @ w2[e]
      gemm(moeHb, w2b + (long long)j * E_ * F_ * D_, slotOut,
           2 * T_, D_, F_, F_, F_, D_,
           0, (long long)F_ * D_, 0, E_, 0, 1.0f,
           counts, offsets, nullptr, slotIds);
      // shared expert
      gemm(xnb, gub + (long long)j * D_ * 2 * F_, gu, T_, 2 * F_, D_,
           D_, D_, 2 * F_, 0, 0, 0, 1, 0, 1.0f,
           nullptr, nullptr, nullptr, nullptr);
      k_sharedact<<<dim3(4096), dim3(256), 0, stream>>>(gu, sact);
      gemm(sact, downb + (long long)j * F_ * D_, sharedTmp, T_, D_, F_,
           F_, F_, D_, 0, 0, 0, 1, 0, 1.0f,
           nullptr, nullptr, nullptr, nullptr);
      k_sgate<<<dim3((T_ + 255) / 256), dim3(256), 0, stream>>>(
          xn, gatew + (long long)j * D_, sg);
      // x += routed + gated shared
      k_combine<<<dim3(4096), dim3(256), 0, stream>>>(x, tw, slotOut, sharedTmp, sg);
    }
  }

  // final rms + lm head
  k_rmsnorm<<<dim3(T_), dim3(256), 0, stream>>>(x, xn, xnb);
  gemm(xnb, lmb, (float*)d_out, T_, V_, D_, D_, D_, V_,
       0, 0, 0, 1, 0, 1.0f, nullptr, nullptr, nullptr, nullptr);
}
